// GroupedQueryAttention_50972671869428
// MI455X (gfx1250) — matrix operand order repaired and checked by reading, not compiled, not run on hardware
//
#include <hip/hip_runtime.h>
#include <hip/hip_bf16.h>

typedef __attribute__((ext_vector_type(16))) __bf16 v16bf;
typedef __attribute__((ext_vector_type(8)))  __bf16 v8bf;
typedef __attribute__((ext_vector_type(2)))  __bf16 v2bf;
typedef __attribute__((ext_vector_type(8)))  float  v8f;

// Explicit global-address-space pointers: guarantees global_load/global_store
// (flat ops tick both LOADcnt and DScnt and serialize against LDS traffic).
#define GAS __attribute__((address_space(1)))
typedef const GAS float* gcfloat;
typedef GAS float*       gfloat;

#define TM 32
#define TN 64
#define TK 32
#define LPAD 40   // LDS row length in bf16 (80 bytes: multiple of 16B for b128 chunks)

struct GemmParams {
  const float* A;      // logical [M x K], K-contiguous (stride 1 along K)
  const float* Bm;     // logical [K x N], generic strides
  const float* bias;   // [N] or nullptr (indexed by global n)
  float* C;            // logical [M x N], col stride 1
  long long aM;        // element stride of A along M
  long long bK, bN;    // element strides of B along K / N
  long long cM;        // row stride of C
  long long aBs, aHs, aKVs;  // per-batch / per-head / per-kv-head offsets
  long long bBs, bHs, bKVs;
  long long cBs, cHs, cKVs;
  int K;               // reduction length (multiple of TK)
  int H, G;            // z = b*H + h ; kv = h / G
  float alpha;         // scale applied to A*B product (bias added after)
};

// Generic batched GEMM: C = alpha * (A @ B) + bias, bf16 WMMA with fp32 accum.
// Block tile 32x64, 8 waves, each wave owns one 16x16 WMMA tile.
__global__ __launch_bounds__(256) void gqa_gemm_wmma(GemmParams p) {
  __shared__ alignas(16) __bf16 lA[TM][LPAD];   // A tile, row-major [m][k]
  __shared__ alignas(16) __bf16 lBt[TN][LPAD];  // B tile, TRANSPOSED [n][k]

  const int tid  = threadIdx.x;
  const int lane = tid & 31;
  const int wave = tid >> 5;   // 0..7
  const int wr   = wave & 1;   // 2 row tiles  (2*16 = 32)
  const int wc   = wave >> 1;  // 4 col tiles  (4*16 = 64)

  const int z  = blockIdx.z;
  const int b  = z / p.H;
  const int h  = z % p.H;
  const int kv = h / p.G;

  const long long offA = (long long)b * p.aBs + (long long)h * p.aHs + (long long)kv * p.aKVs;
  const long long offB = (long long)b * p.bBs + (long long)h * p.bHs + (long long)kv * p.bKVs;
  const long long offC = (long long)b * p.cBs + (long long)h * p.cHs + (long long)kv * p.cKVs;

  const int m0 = blockIdx.x * TM;
  const int n0 = blockIdx.y * TN;

  // ---- Hoisted staging addresses (advance by a constant each K step) ----
  // A tile: 32x32 = 1024 elems; each thread stages 2 pairs of adjacent K elems.
  gcfloat asrc[2];
  __bf16* adst[2];
  {
    gcfloat Ab = (gcfloat)(p.A) + offA;
    #pragma unroll
    for (int j = 0; j < 2; ++j) {
      const int i = tid * 2 + j * 512;       // even, covers 0..1023
      const int r = i >> 5, c = i & 31;      // c even
      asrc[j] = Ab + (long long)(m0 + r) * p.aM + c;
      adst[j] = &lA[r][c];
    }
  }
  // B tile: 32x64 = 2048 elems; each thread stages 8 scalars into lBt[n][k].
  gcfloat bsrc[8];
  __bf16* bdst[8];
  {
    gcfloat Bb = (gcfloat)(p.Bm) + offB;
    #pragma unroll
    for (int j = 0; j < 8; ++j) {
      const int i = tid + j * 256;
      const int kr = i >> 6, nc = i & 63;
      bsrc[j] = Bb + (long long)kr * p.bK + (long long)(n0 + nc) * p.bN;
      bdst[j] = &lBt[nc][kr];
    }
  }
  const long long aAdv = TK;                 // A is K-contiguous
  const long long bAdv = (long long)TK * p.bK;

  // ---- Hoisted fragment addresses (constant; tile contents change per step) ----
  // A 16x32 bf16: lanes 0-15 row M=lane hold K 0..7 & 16..23; lanes 16-31 hold K 8..15 & 24..31.
  const int ar  = wr * 16 + (lane & 15);
  const int ach = (lane & 16) ? 1 : 0;                 // 8-elem chunk: K 0..7 or 8..15
  const v8bf* ap = reinterpret_cast<const v8bf*>(&lA[ar][0]);
  // B 32x16 bf16: lanes 0-15 col N=lane hold K 0..7 & 16..23; lanes 16-31 hold K 8..15 & 24..31.
  const int bn  = wc * 16 + (lane & 15);
  const int bch = (lane & 16) ? 1 : 0;                 // 8-elem chunk: K 0..7 or 8..15
  const v8bf* bp = reinterpret_cast<const v8bf*>(&lBt[bn][0]);

  v8f acc;
  #pragma unroll
  for (int i = 0; i < 8; ++i) acc[i] = 0.0f;

  for (int k0 = 0; k0 < p.K; k0 += TK) {
    // Phase 1: issue ALL global loads (distinct registers -> overlapped latency).
    float ax0[2], ax1[2], bx[8];
    #pragma unroll
    for (int j = 0; j < 2; ++j) {
      ax0[j] = asrc[j][0];
      ax1[j] = asrc[j][1];
      asrc[j] += aAdv;
    }
    #pragma unroll
    for (int j = 0; j < 8; ++j) {
      bx[j] = *bsrc[j];
      bsrc[j] += bAdv;
    }
    // Phase 2: convert fp32 -> bf16 and stage into LDS.
    #pragma unroll
    for (int j = 0; j < 2; ++j) {
      *reinterpret_cast<v2bf*>(adst[j]) = (v2bf){(__bf16)ax0[j], (__bf16)ax1[j]};
    }
    #pragma unroll
    for (int j = 0; j < 8; ++j) {
      *bdst[j] = (__bf16)bx[j];
    }
    __syncthreads();

    const v8bf a_lo = ap[ach];
    const v8bf a_hi = ap[2 + ach];
    const v16bf afrag = __builtin_shufflevector(a_lo, a_hi,
        0, 1, 2, 3, 4, 5, 6, 7, 8, 9, 10, 11, 12, 13, 14, 15);
    const v8bf b_lo = bp[bch];
    const v8bf b_hi = bp[2 + bch];
    const v16bf bfrag = __builtin_shufflevector(b_lo, b_hi,
        0, 1, 2, 3, 4, 5, 6, 7, 8, 9, 10, 11, 12, 13, 14, 15);

    acc = __builtin_amdgcn_wmma_f32_16x16x32_bf16(false, afrag, false, bfrag,
                                                  (short)0, acc, false, false);
    __syncthreads();
  }

  // C 16x16 f32: VGPR v -> M = v (lanes 0-15) / v+8 (lanes 16-31), N = lane&15.
  const int gn = n0 + wc * 16 + (lane & 15);
  const int mb = (lane & 16) ? 8 : 0;
  const float bv = p.bias ? ((gcfloat)(p.bias))[gn] : 0.0f;
  gfloat crow = (gfloat)(p.C) + offC + (long long)(m0 + wr * 16 + mb) * p.cM + gn;
  #pragma unroll
  for (int v = 0; v < 8; ++v) {
    *crow = acc[v] * p.alpha + bv;
    crow += p.cM;
  }
}

// In-place row softmax: one 256-thread block per row of `ncols` floats.
__global__ __launch_bounds__(256) void gqa_softmax_rows(float* __restrict__ attn, int ncols) {
  __shared__ float red[256];
  gfloat p = (gfloat)(attn) + (long long)blockIdx.x * (long long)ncols;
  const int tid = threadIdx.x;

  float m = -3.4e38f;
  for (int i = tid; i < ncols; i += 256) m = fmaxf(m, p[i]);
  red[tid] = m;
  __syncthreads();
  for (int s = 128; s > 0; s >>= 1) {
    if (tid < s) red[tid] = fmaxf(red[tid], red[tid + s]);
    __syncthreads();
  }
  m = red[0];
  __syncthreads();

  float sum = 0.0f;
  for (int i = tid; i < ncols; i += 256) {
    const float e = __expf(p[i] - m);
    p[i] = e;
    sum += e;
  }
  red[tid] = sum;
  __syncthreads();
  for (int s = 128; s > 0; s >>= 1) {
    if (tid < s) red[tid] += red[tid + s];
    __syncthreads();
  }
  const float inv = 1.0f / red[0];
  for (int i = tid; i < ncols; i += 256) p[i] *= inv;
}

extern "C" void kernel_launch(void* const* d_in, const int* in_sizes, int n_in,
                              void* d_out, int out_size, void* d_ws, size_t ws_size,
                              hipStream_t stream) {
  (void)in_sizes; (void)n_in; (void)out_size; (void)ws_size;

  constexpr int B = 2, S = 2048, D = 1024, H = 16, KV = 4, G = 4, DK = 64;
  constexpr long long MS = (long long)B * S;   // 4096 token rows
  constexpr int KVD = KV * DK;                 // 256

  const float* query = (const float*)d_in[0];
  const float* key   = (const float*)d_in[1];
  const float* value = (const float*)d_in[2];
  // d_in[3] = mask (int32): intentionally unused (reference's masked_fill is a no-op)
  const float* w_q = (const float*)d_in[4];
  const float* b_q = (const float*)d_in[5];
  const float* w_k = (const float*)d_in[6];
  const float* b_k = (const float*)d_in[7];
  const float* w_v = (const float*)d_in[8];
  const float* b_v = (const float*)d_in[9];
  const float* w_o = (const float*)d_in[10];
  const float* b_o = (const float*)d_in[11];

  float* out  = (float*)d_out;                       // [B,S,D]
  float* attn = out + (long long)B * S * D;          // [B,H,S,S] (raw scores -> softmaxed in place)

  float* Qbuf = (float*)d_ws;                        // [B*S, D]
  float* Kbuf = Qbuf + MS * D;                       // [B*S, KVD]
  float* Vbuf = Kbuf + MS * KVD;                     // [B*S, KVD]
  float* ctx  = Vbuf + MS * KVD;                     // [B*S, D]

  const dim3 blk(256);

  // ---- Q projection: Qbuf = query @ w_q + b_q  (4096 x 1024 x 1024) ----
  {
    GemmParams p{};
    p.A = query; p.Bm = w_q; p.bias = b_q; p.C = Qbuf;
    p.aM = D; p.bK = D; p.bN = 1; p.cM = D;
    p.K = D; p.H = 1; p.G = 1; p.alpha = 1.0f;
    gqa_gemm_wmma<<<dim3(MS / TM, D / TN, 1), blk, 0, stream>>>(p);
  }
  // ---- K projection: Kbuf = key @ w_k + b_k  (4096 x 256 x 1024) ----
  {
    GemmParams p{};
    p.A = key; p.Bm = w_k; p.bias = b_k; p.C = Kbuf;
    p.aM = D; p.bK = KVD; p.bN = 1; p.cM = KVD;
    p.K = D; p.H = 1; p.G = 1; p.alpha = 1.0f;
    gqa_gemm_wmma<<<dim3(MS / TM, KVD / TN, 1), blk, 0, stream>>>(p);
  }
  // ---- V projection: Vbuf = value @ w_v + b_v ----
  {
    GemmParams p{};
    p.A = value; p.Bm = w_v; p.bias = b_v; p.C = Vbuf;
    p.aM = D; p.bK = KVD; p.bN = 1; p.cM = KVD;
    p.K = D; p.H = 1; p.G = 1; p.alpha = 1.0f;
    gqa_gemm_wmma<<<dim3(MS / TM, KVD / TN, 1), blk, 0, stream>>>(p);
  }
  // ---- Scores: attn[b,h,sq,sk] = scale * sum_d Q[b,sq,h*64+d] * K[b,sk,kv*64+d] ----
  {
    GemmParams p{};
    p.A = Qbuf; p.Bm = Kbuf; p.bias = nullptr; p.C = attn;
    p.aM = D;
    p.bK = 1; p.bN = KVD;                 // B[k][n] = Kbuf[(b*S+n)*256 + kv*64 + k]
    p.cM = S;
    p.aBs = (long long)S * D;   p.aHs = DK;               p.aKVs = 0;
    p.bBs = (long long)S * KVD; p.bHs = 0;                p.bKVs = DK;
    p.cBs = (long long)H * S * S; p.cHs = (long long)S * S; p.cKVs = 0;
    p.K = DK; p.H = H; p.G = G; p.alpha = 0.125f;         // 1/sqrt(DK)
    gqa_gemm_wmma<<<dim3(S / TM, S / TN, B * H), blk, 0, stream>>>(p);
  }
  // ---- Softmax over the last axis, in place (B*H*S rows of S) ----
  gqa_softmax_rows<<<dim3((unsigned)(B * H * S)), blk, 0, stream>>>(attn, S);

  // ---- Context: ctx[b,sq,h*64+d] = sum_sk attn[b,h,sq,sk] * V[b,sk,kv*64+d] ----
  {
    GemmParams p{};
    p.A = attn; p.Bm = Vbuf; p.bias = nullptr; p.C = ctx;
    p.aM = S;
    p.bK = KVD; p.bN = 1;                 // B[k][n] = Vbuf[(b*S+k)*256 + kv*64 + n]
    p.cM = D;
    p.aBs = (long long)H * S * S; p.aHs = (long long)S * S; p.aKVs = 0;
    p.bBs = (long long)S * KVD;   p.bHs = 0;                p.bKVs = DK;
    p.cBs = (long long)S * D;     p.cHs = DK;               p.cKVs = 0;
    p.K = S; p.H = H; p.G = G; p.alpha = 1.0f;
    gqa_gemm_wmma<<<dim3(S / TM, DK / TN, B * H), blk, 0, stream>>>(p);
  }
  // ---- Output projection: out = ctx @ w_o + b_o  (4096 x 1024 x 1024) ----
  {
    GemmParams p{};
    p.A = ctx; p.Bm = w_o; p.bias = b_o; p.C = out;
    p.aM = D; p.bK = D; p.bN = 1; p.cM = D;
    p.K = D; p.H = 1; p.G = 1; p.alpha = 1.0f;
    gqa_gemm_wmma<<<dim3(MS / TM, D / TN, 1), blk, 0, stream>>>(p);
  }
}
